// LocalTransformerBlock_6210522710229
// MI455X (gfx1250) — matrix operand order repaired and checked by reading, not compiled, not run on hardware
//
#include <hip/hip_runtime.h>
#include <hip/hip_bf16.h>
#include <math.h>

// ---------------- CDNA5 WMMA types ----------------
typedef __attribute__((ext_vector_type(16))) _Float16 v16h;
typedef __attribute__((ext_vector_type(8)))  float    v8f;
typedef __attribute__((ext_vector_type(4)))  int      v4i;

union Frag16 { v16h v; uint4 q[2]; };

__device__ inline v16h lds_load16(const _Float16* p) {
  Frag16 f;
  f.q[0] = *(const uint4*)(p);
  f.q[1] = *(const uint4*)(p + 8);
  return f.v;
}

// ---------------- CDNA5 async global->LDS copy (verified on this toolchain) ----------------
#if defined(__has_builtin)
#if __has_builtin(__builtin_amdgcn_global_load_async_to_lds_b128)
#define HAVE_ASYNC_LDS 1
#endif
#endif

typedef __attribute__((address_space(1))) v4i* as1_v4i;
typedef __attribute__((address_space(3))) v4i* as3_v4i;

__device__ inline void copy16_g2lds(const void* g, void* lds) {
#ifdef HAVE_ASYNC_LDS
  __builtin_amdgcn_global_load_async_to_lds_b128(
      (as1_v4i)(unsigned long long)(uintptr_t)g,
      (as3_v4i)(unsigned int)(uintptr_t)lds,
      0, 0);
#else
  *(uint4*)lds = *(const uint4*)g;
#endif
}

__device__ inline void async_wait0() {
#ifdef HAVE_ASYNC_LDS
#if __has_builtin(__builtin_amdgcn_s_wait_asynccnt)
  __builtin_amdgcn_s_wait_asynccnt(0);
#else
  asm volatile("s_wait_asynccnt 0x0" ::: "memory");
#endif
#endif
}

// ---------------- problem constants ----------------
constexpr int BB    = 2;
constexpr int NN    = 8192;
constexpr int DIM   = 1024;
constexpr int HEADS = 8;
constexpr int DHEAD = 64;
constexpr int INNER = 512;     // HEADS * DHEAD
constexpr int WIN   = 512;
constexpr int HID   = 2730;    // int(1024*4*2/3)
constexpr int ALDA  = 2736;    // HID padded to multiple of 8 (16B rows)
constexpr int MROWS = BB * NN; // 16384
constexpr float QK_SCALE = 8.0f;

// =====================================================================
// Tiled WMMA GEMM with double-buffered LDS + async-to-LDS pipelining.
// Out[M,Ncols] (+= if mode==1) A[M,K](f16,lda) * W[K,Ncols](f32->f16) (+bias)
// 128x128 block tile, 256 threads = 8 waves (2x4), each wave 64x32 (4x2 WMMA).
// Pipeline per K-tile: async-A(k+1) -> B-loads(k+1)->regs -> WMMA(k)
//                      -> B-regs->LDS -> wait_async -> barrier -> swap.
// =====================================================================
__global__ __launch_bounds__(256) void gemm_kernel(
    const _Float16* __restrict__ A,
    const float*    __restrict__ W,
    const float*    __restrict__ bias,
    float*          __restrict__ Out,
    int M, int K, int lda, int Ncols, int mode)
{
  __shared__ _Float16 Alds[2][128 * 32]; // [m][k]
  __shared__ _Float16 Blds[2][128 * 32]; // [n][k] (transposed stage)

  int m0  = blockIdx.x * 128;
  int n0  = blockIdx.y * 128;
  int tid = threadIdx.x;
  int wv = tid >> 5, lane = tid & 31;
  int l16 = lane & 15, lh = lane >> 4;
  int wm0 = (wv >> 2) * 64;
  int wn0 = (wv & 3) * 32;

  // ---- staging helpers ----
  auto stageA_fast = [&](int buf, int k0) {
#pragma unroll
    for (int i = 0; i < 2; ++i) {
      int idx = i * 256 + tid;
      int mm = idx >> 2, kk8 = (idx & 3) * 8;
      copy16_g2lds(&A[(size_t)(m0 + mm) * lda + k0 + kk8], &Alds[buf][mm * 32 + kk8]);
    }
  };
  auto stageA_slow = [&](int buf, int k0) {
#pragma unroll
    for (int i = 0; i < 16; ++i) {
      int idx = i * 256 + tid;
      int kk = idx & 31, mm = idx >> 5;
      int kg = k0 + kk;
      _Float16 val = (_Float16)0.f;
      if (kg < K) val = A[(size_t)(m0 + mm) * lda + kg];
      Alds[buf][mm * 32 + kk] = val;
    }
  };
  auto stageB_slow = [&](int buf, int k0) {
#pragma unroll
    for (int i = 0; i < 16; ++i) {
      int idx = i * 256 + tid;
      int nn = idx & 127, kk = idx >> 7;
      int kg = k0 + kk, ng = n0 + nn;
      float v = 0.f;
      if (kg < K && ng < Ncols) v = W[(size_t)kg * Ncols + ng];
      Blds[buf][nn * 32 + kk] = (_Float16)v;
    }
  };

  v8f acc[4][2] = {};

  // ---- prologue: stage tile 0 ----
  {
    bool fullK = (32 <= K);
    if (fullK) stageA_fast(0, 0); else stageA_slow(0, 0);
    if (fullK && (n0 + 128 <= Ncols)) {
#pragma unroll
      for (int i = 0; i < 4; ++i) {
        int idx = i * 256 + tid;
        int kk = idx >> 5, nn4 = (idx & 31) * 4;
        const float4 w4 = *(const float4*)&W[(size_t)kk * Ncols + n0 + nn4];
        Blds[0][(nn4 + 0) * 32 + kk] = (_Float16)w4.x;
        Blds[0][(nn4 + 1) * 32 + kk] = (_Float16)w4.y;
        Blds[0][(nn4 + 2) * 32 + kk] = (_Float16)w4.z;
        Blds[0][(nn4 + 3) * 32 + kk] = (_Float16)w4.w;
      }
    } else {
      stageB_slow(0, 0);
    }
    async_wait0();
    __syncthreads();
  }

  int p = 0;
  for (int k0 = 0; k0 < K; k0 += 32) {
    int kn = k0 + 32;
    bool nxt      = kn < K;
    bool nxtFullK = (kn + 32 <= K);
    bool fastB    = nxt && nxtFullK && (n0 + 128 <= Ncols);

    // 1) issue async DMA of next A tile (overlaps with WMMA below)
    if (nxt && nxtFullK) stageA_fast(p ^ 1, kn);

    // 2) issue next B tile global loads into registers
    float4 wreg[4];
    if (fastB) {
#pragma unroll
      for (int i = 0; i < 4; ++i) {
        int idx = i * 256 + tid;
        int kk = idx >> 5, nn4 = (idx & 31) * 4;
        wreg[i] = *(const float4*)&W[(size_t)(kn + kk) * Ncols + n0 + nn4];
      }
    }
    // prefetch tile k+2 of W (global_prefetch_b8)
    if (kn + 32 < K) {
      __builtin_prefetch(&W[(size_t)(kn + 32 + (tid & 31)) * Ncols + n0], 0, 3);
    }

    // 3) compute tile k from buffer p
    {
      const _Float16* Bl = Blds[p];
      const _Float16* Al = Alds[p];
      Frag16 bf0, bf1;
      bf0.v = lds_load16(&Bl[(wn0 + l16) * 32 + lh * 16]);
      bf1.v = lds_load16(&Bl[(wn0 + 16 + l16) * 32 + lh * 16]);
#pragma unroll
      for (int i = 0; i < 4; ++i) {
        Frag16 af;
        int row = wm0 + i * 16 + l16;
        int base = lh * 16;
        af.q[0] = *(const uint4*)&Al[row * 32 + base];
        af.q[1] = *(const uint4*)&Al[row * 32 + base + 8];
        acc[i][0] = __builtin_amdgcn_wmma_f32_16x16x32_f16(false, af.v, false, bf0.v,
                                                           (short)0, acc[i][0], false, false);
        acc[i][1] = __builtin_amdgcn_wmma_f32_16x16x32_f16(false, af.v, false, bf1.v,
                                                           (short)0, acc[i][1], false, false);
      }
    }

    // 4) finish staging next tile into buffer p^1
    if (nxt) {
      if (fastB) {
#pragma unroll
        for (int i = 0; i < 4; ++i) {
          int idx = i * 256 + tid;
          int kk = idx >> 5, nn4 = (idx & 31) * 4;
          Blds[p ^ 1][(nn4 + 0) * 32 + kk] = (_Float16)wreg[i].x;
          Blds[p ^ 1][(nn4 + 1) * 32 + kk] = (_Float16)wreg[i].y;
          Blds[p ^ 1][(nn4 + 2) * 32 + kk] = (_Float16)wreg[i].z;
          Blds[p ^ 1][(nn4 + 3) * 32 + kk] = (_Float16)wreg[i].w;
        }
      } else {
        if (!nxtFullK) stageA_slow(p ^ 1, kn);
        stageB_slow(p ^ 1, kn);
      }
    }
    async_wait0();
    __syncthreads();
    p ^= 1;
  }

  // epilogue: D layout lane l -> col l16, row r + 8*lh
#pragma unroll
  for (int i = 0; i < 4; ++i) {
#pragma unroll
    for (int j = 0; j < 2; ++j) {
      int col = n0 + wn0 + j * 16 + l16;
      if (col < Ncols) {
#pragma unroll
        for (int r = 0; r < 8; ++r) {
          int row = m0 + wm0 + i * 16 + r + 8 * lh;
          size_t off = (size_t)row * Ncols + col;
          float v = acc[i][j][r];
          if (bias) v += bias[col];
          if (mode == 1) v += Out[off];
          Out[off] = v;
        }
      }
    }
  }
}

// =====================================================================
// shift_token(x) cast to f16 (second half of channels comes from token n-1)
// vectorized: 4 elements per thread
// =====================================================================
__global__ void shift_cast_kernel(const float* __restrict__ x, _Float16* __restrict__ out)
{
  size_t e = ((size_t)blockIdx.x * 256 + threadIdx.x) * 4; // B*N*DIM / 4 threads
  int d = (int)(e & (DIM - 1));
  int n = (int)((e >> 10) & (NN - 1));
  float4 v;
  if (d < DIM / 2) {
    v = *(const float4*)&x[e];
  } else if (n == 0) {
    v = make_float4(0.f, 0.f, 0.f, 0.f);
  } else {
    v = *(const float4*)&x[e - DIM];
  }
  uint2 pk;
  _Float16* ph = (_Float16*)&pk;
  ph[0] = (_Float16)v.x; ph[1] = (_Float16)v.y;
  ph[2] = (_Float16)v.z; ph[3] = (_Float16)v.w;
  *(uint2*)&out[e] = pk;
}

// =====================================================================
// qk prep: l2norm * scale, then rotary-xpos, in place on qkv f32 buffer.
// one thread per (b, n, head)
// =====================================================================
__global__ void qkprep_kernel(float* __restrict__ qkv,
                              const float* __restrict__ qs,
                              const float* __restrict__ ks)
{
  int idx = blockIdx.x * 256 + threadIdx.x; // MROWS*HEADS exactly
  int h = idx & 7;
  int row = idx >> 3;          // b*NN + n
  int n = row & (NN - 1);
  float* qp = qkv + (size_t)row * (3 * INNER) + h * DHEAD;
  float* kp = qp + INNER;

  float q[64], k[64];
  float qn = 0.f, kn = 0.f;
#pragma unroll 4
  for (int i = 0; i < 16; ++i) {
    float4 q4 = *(const float4*)&qp[i * 4];
    float4 k4 = *(const float4*)&kp[i * 4];
    q[i * 4 + 0] = q4.x; q[i * 4 + 1] = q4.y; q[i * 4 + 2] = q4.z; q[i * 4 + 3] = q4.w;
    k[i * 4 + 0] = k4.x; k[i * 4 + 1] = k4.y; k[i * 4 + 2] = k4.z; k[i * 4 + 3] = k4.w;
    qn += q4.x*q4.x + q4.y*q4.y + q4.z*q4.z + q4.w*q4.w;
    kn += k4.x*k4.x + k4.y*k4.y + k4.z*k4.z + k4.w*k4.w;
  }
  qn = fmaxf(sqrtf(qn), 1e-12f);
  kn = fmaxf(sqrtf(kn), 1e-12f);
  float pw = ((float)n - 4096.f) / 256.f; // (n - N//2) / (WINDOW//2)

#pragma unroll 4
  for (int i = 0; i < 32; ++i) {
    float invf = powf(10000.f, -(2.f * i) / 64.f);
    float ang = (float)n * invf;
    float cv = cosf(ang), sv = sinf(ang);
    float sc = powf((2.f * i + 25.6f) / 89.6f, pw);
    float isc = 1.f / sc;
    float q1 = q[i] / qn * qs[i];
    float q2 = q[i + 32] / qn * qs[i + 32];
    float k1 = k[i] / kn * ks[i];
    float k2 = k[i + 32] / kn * ks[i + 32];
    qp[i]      = (q1 * cv - q2 * sv) * sc;
    qp[i + 32] = (q2 * cv + q1 * sv) * sc;
    kp[i]      = (k1 * cv - k2 * sv) * isc;
    kp[i + 32] = (k2 * cv + k1 * sv) * isc;
  }
}

// =====================================================================
// Local windowed attention (flash-style) with WMMA.
// block = (b, head, window, quarter): 256 threads = 8 waves, 16 q-rows/wave.
// Streams the 1024-row look-around K window in 128-row LDS chunks
// (only the 5 chunks overlapping the valid mask).
// Writes output directly as f16 (A input for the wo GEMM).
// =====================================================================
__global__ __launch_bounds__(256) void attn_kernel(const float* __restrict__ qkv,
                                                   _Float16* __restrict__ o16)
{
  __shared__ _Float16 Klds[128 * 64];   // [krow][d]
  __shared__ _Float16 Vlds[64 * 128];   // [d][krow] (transposed)
  __shared__ _Float16 Plds[8][16 * 32]; // per-wave P tile

  int blk = blockIdx.x;
  int qq = blk & 3;
  int w  = (blk >> 2) & 15;
  int h  = (blk >> 6) & 7;
  int b  = blk >> 9;
  int i0 = w * WIN + qq * 128;

  int tid = threadIdx.x;
  int wv = tid >> 5, lane = tid & 31;
  int l16 = lane & 15, lh = lane >> 4;
  int qrow = i0 + wv * 16;

  // Q fragments (16x64 as two 16x32 A-frags), scaled by QK_SCALE
  Frag16 qa[2];
  {
    const float* qp = qkv + ((size_t)(b * NN + qrow + l16)) * (3 * INNER) + h * DHEAD;
    int base = lh * 16;
#pragma unroll
    for (int half = 0; half < 2; ++half) {
#pragma unroll
      for (int j = 0; j < 8; ++j) {
        qa[half].v[j]     = (_Float16)(qp[half * 32 + base + j] * QK_SCALE);
        qa[half].v[8 + j] = (_Float16)(qp[half * 32 + base + 8 + j] * QK_SCALE);
      }
    }
  }

  v8f oacc[4] = {};
  float mrow[8], lrow[8];
#pragma unroll
  for (int r = 0; r < 8; ++r) { mrow[r] = -1e30f; lrow[r] = 0.f; }

  for (int c = 0; c < 5; ++c) {
    int jb = i0 - WIN + c * 128; // chunk base token (may be negative)
    __syncthreads();
    // stage K (row-major) and V (transposed) chunk, f32 -> f16, float4 vectorized
#pragma unroll
    for (int i = 0; i < 8; ++i) {
      int idx = i * 256 + tid; // 2048 float4 groups
      int dd4 = (idx & 15) * 4, rr = idx >> 4;
      int jrow = jb + rr;
      float4 k4 = make_float4(0.f, 0.f, 0.f, 0.f);
      float4 v4 = make_float4(0.f, 0.f, 0.f, 0.f);
      if (jrow >= 0) {
        const float* bp = qkv + ((size_t)(b * NN + jrow)) * (3 * INNER) + h * DHEAD;
        k4 = *(const float4*)&bp[INNER + dd4];
        v4 = *(const float4*)&bp[2 * INNER + dd4];
      }
      // K row-major: 4 contiguous halves -> one 8B store
      uint2 pk;
      _Float16* ph = (_Float16*)&pk;
      ph[0] = (_Float16)k4.x; ph[1] = (_Float16)k4.y;
      ph[2] = (_Float16)k4.z; ph[3] = (_Float16)k4.w;
      *(uint2*)&Klds[rr * 64 + dd4] = pk;
      // V transposed: strided scalar stores
      Vlds[(dd4 + 0) * 128 + rr] = (_Float16)v4.x;
      Vlds[(dd4 + 1) * 128 + rr] = (_Float16)v4.y;
      Vlds[(dd4 + 2) * 128 + rr] = (_Float16)v4.z;
      Vlds[(dd4 + 3) * 128 + rr] = (_Float16)v4.w;
    }
    __syncthreads();

    _Float16* P = &Plds[wv][0];
    for (int jj = 0; jj < 128; jj += 32) {
      // sim tiles: s0 = cols jj..jj+15, s1 = cols jj+16..jj+31
      v8f s0 = {}, s1 = {};
      {
        v16h kb;
        kb = lds_load16(&Klds[(jj + l16) * 64 + lh * 16]);
        s0 = __builtin_amdgcn_wmma_f32_16x16x32_f16(false, qa[0].v, false, kb, (short)0, s0, false, false);
        kb = lds_load16(&Klds[(jj + l16) * 64 + 32 + lh * 16]);
        s0 = __builtin_amdgcn_wmma_f32_16x16x32_f16(false, qa[1].v, false, kb, (short)0, s0, false, false);
        kb = lds_load16(&Klds[(jj + 16 + l16) * 64 + lh * 16]);
        s1 = __builtin_amdgcn_wmma_f32_16x16x32_f16(false, qa[0].v, false, kb, (short)0, s1, false, false);
        kb = lds_load16(&Klds[(jj + 16 + l16) * 64 + 32 + lh * 16]);
        s1 = __builtin_amdgcn_wmma_f32_16x16x32_f16(false, qa[1].v, false, kb, (short)0, s1, false, false);
      }
      int jt0 = jb + jj + l16;
      int jt1 = jt0 + 16;
#pragma unroll
      for (int r = 0; r < 8; ++r) {
        int irow = qrow + r + 8 * lh;
        if (jt0 < 0 || jt0 > irow || jt0 + WIN < irow) s0[r] = -1e30f;
        if (jt1 < 0 || jt1 > irow || jt1 + WIN < irow) s1[r] = -1e30f;
        float mx = fmaxf(s0[r], s1[r]);
#pragma unroll
        for (int d = 1; d < 16; d <<= 1) mx = fmaxf(mx, __shfl_xor(mx, d, 32));
        float mn = fmaxf(mrow[r], mx);
        float alpha = expf(mrow[r] - mn);
        lrow[r] *= alpha;
#pragma unroll
        for (int nd = 0; nd < 4; ++nd) oacc[nd][r] *= alpha;
        float p0 = expf(s0[r] - mn);
        float p1 = expf(s1[r] - mn);
        float ps = p0 + p1;
#pragma unroll
        for (int d = 1; d < 16; d <<= 1) ps += __shfl_xor(ps, d, 32);
        lrow[r] += ps;
        mrow[r] = mn;
        // store P in row-major [row][col] f16 (wave-private LDS)
        P[(r + 8 * lh) * 32 + l16]      = (_Float16)p0;
        P[(r + 8 * lh) * 32 + 16 + l16] = (_Float16)p1;
      }
      // intra-wave LDS RAW: wait for DS stores before A-frag reads
      asm volatile("s_wait_dscnt 0x0" ::: "memory");
      Frag16 pa;
      {
        int base = lh * 16;
        pa.q[0] = *(const uint4*)&P[l16 * 32 + base];
        pa.q[1] = *(const uint4*)&P[l16 * 32 + base + 8];
      }
      // out += P(16x32) x V(32x64)
#pragma unroll
      for (int nd = 0; nd < 4; ++nd) {
        v16h vb = lds_load16(&Vlds[(nd * 16 + l16) * 128 + jj + lh * 16]);
        oacc[nd] = __builtin_amdgcn_wmma_f32_16x16x32_f16(false, pa.v, false, vb,
                                                          (short)0, oacc[nd], false, false);
      }
    }
  }

  // normalize and write f16 output (B,N,INNER)
#pragma unroll
  for (int nd = 0; nd < 4; ++nd) {
#pragma unroll
    for (int r = 0; r < 8; ++r) {
      int irow = qrow + r + 8 * lh;
      float val = oacc[nd][r] / lrow[r];
      o16[((size_t)(b * NN + irow)) * INNER + h * DHEAD + nd * 16 + l16] = (_Float16)val;
    }
  }
}

// =====================================================================
// shift_token + LayerNorm -> f16  (one wave per row)
// =====================================================================
__global__ __launch_bounds__(256) void lnshift_kernel(const float* __restrict__ x,
                                                      const float* __restrict__ g,
                                                      const float* __restrict__ bta,
                                                      _Float16* __restrict__ out)
{
  int row = blockIdx.x * 8 + (threadIdx.x >> 5);
  int lane = threadIdx.x & 31;
  int n = row & (NN - 1);
  float hvals[32];
  float s = 0.f, s2 = 0.f;
#pragma unroll
  for (int i = 0; i < 32; ++i) {
    int d = i * 32 + lane;
    float v;
    if (d < DIM / 2) v = x[(size_t)row * DIM + d];
    else             v = (n == 0) ? 0.f : x[(size_t)(row - 1) * DIM + d];
    hvals[i] = v; s += v; s2 += v * v;
  }
#pragma unroll
  for (int d = 1; d < 32; d <<= 1) { s += __shfl_xor(s, d, 32); s2 += __shfl_xor(s2, d, 32); }
  float mu = s * (1.f / DIM);
  float var = s2 * (1.f / DIM) - mu * mu;
  float rstd = rsqrtf(var + 1e-5f);
#pragma unroll
  for (int i = 0; i < 32; ++i) {
    int d = i * 32 + lane;
    out[(size_t)row * DIM + d] = (_Float16)((hvals[i] - mu) * rstd * g[d] + bta[d]);
  }
}

// =====================================================================
// GEGLU: a * gelu(g) (exact erf), f32 hidden -> f16 activation (row stride ALDA)
// =====================================================================
__global__ void geglu_kernel(const float* __restrict__ hid, _Float16* __restrict__ out)
{
  size_t idx = (size_t)blockIdx.x * 256 + threadIdx.x;
  if (idx >= (size_t)MROWS * HID) return;
  size_t row = idx / HID;
  int j = (int)(idx - row * HID);
  float a  = hid[row * (size_t)(2 * HID) + j];
  float gg = hid[row * (size_t)(2 * HID) + HID + j];
  float ge = 0.5f * gg * (1.f + erff(gg * 0.70710678118f));
  out[row * (size_t)ALDA + j] = (_Float16)(a * ge);
}

// =====================================================================
// host driver
// =====================================================================
extern "C" void kernel_launch(void* const* d_in, const int* in_sizes, int n_in,
                              void* d_out, int out_size, void* d_ws, size_t ws_size,
                              hipStream_t stream)
{
  (void)in_sizes; (void)n_in; (void)out_size; (void)ws_size;
  const float* x0   = (const float*)d_in[0];
  const float* wqkv = (const float*)d_in[1];
  const float* qsc  = (const float*)d_in[2];
  const float* ksc  = (const float*)d_in[3];
  const float* wo   = (const float*)d_in[4];
  const float* lng  = (const float*)d_in[5];
  const float* lnb  = (const float*)d_in[6];
  const float* w1   = (const float*)d_in[7];
  const float* b1   = (const float*)d_in[8];
  const float* w2   = (const float*)d_in[9];
  const float* b2   = (const float*)d_in[10];

  float* x = (float*)d_out; // x lives in d_out, updated in place
  char* ws = (char*)d_ws;
  _Float16* h16   = (_Float16*)ws;                                          // 32 MB
  _Float16* act16 = (_Float16*)(ws + (size_t)MROWS * DIM * 2);              // 90 MB (attn-out / geglu-out, lda=ALDA)
  float*    tmp   = (float*)(ws + (size_t)MROWS * DIM * 2
                                + (size_t)MROWS * ALDA * 2);                // 340 MB (qkv / ffn hidden)

  (void)hipMemcpyAsync(x, x0, (size_t)MROWS * DIM * sizeof(float),
                       hipMemcpyDeviceToDevice, stream);

  for (int l = 0; l < 2; ++l) {
    // h = shift_token(x) -> f16
    shift_cast_kernel<<<(MROWS * DIM / 4) / 256, 256, 0, stream>>>(x, h16);
    // qkv = h @ wqkv
    gemm_kernel<<<dim3(MROWS / 128, (3 * INNER) / 128), 256, 0, stream>>>(
        h16, wqkv + (size_t)l * DIM * 3 * INNER, nullptr, tmp, MROWS, DIM, DIM, 3 * INNER, 0);
    // q,k: l2norm*scale + rotary-xpos (in place)
    qkprep_kernel<<<(MROWS * HEADS) / 256, 256, 0, stream>>>(
        tmp, qsc + (size_t)l * DHEAD, ksc + (size_t)l * DHEAD);
    // local attention -> act16 (f16, lda=INNER)
    attn_kernel<<<BB * HEADS * (NN / WIN) * 4, 256, 0, stream>>>(tmp, act16);
    // x = x + o @ wo
    gemm_kernel<<<dim3(MROWS / 128, DIM / 128), 256, 0, stream>>>(
        act16, wo + (size_t)l * INNER * DIM, nullptr, x, MROWS, INNER, INNER, DIM, 1);
    // h = LN(shift_token(x)) -> f16
    lnshift_kernel<<<MROWS / 8, 256, 0, stream>>>(
        x, lng + (size_t)l * DIM, lnb + (size_t)l * DIM, h16);
    // hid = h @ w1 + b1
    gemm_kernel<<<dim3(MROWS / 128, (2 * HID + 127) / 128), 256, 0, stream>>>(
        h16, w1 + (size_t)l * DIM * 2 * HID, b1 + (size_t)l * 2 * HID, tmp,
        MROWS, DIM, DIM, 2 * HID, 0);
    // act = a * gelu(g) -> f16 (row stride ALDA)
    geglu_kernel<<<(int)(((size_t)MROWS * HID + 255) / 256), 256, 0, stream>>>(tmp, act16);
    // x = x + act @ w2 + b2
    gemm_kernel<<<dim3(MROWS / 128, DIM / 128), 256, 0, stream>>>(
        act16, w2 + (size_t)l * HID * DIM, b2 + (size_t)l * DIM, x, MROWS, HID, ALDA, DIM, 1);
  }
}
